// gruH1_46351287058869
// MI455X (gfx1250) — matrix operand order repaired and checked by reading, not compiled, not run on hardware
//
#include <hip/hip_runtime.h>
#include <hip/hip_bf16.h>
#include <math.h>

// ---------------------------------------------------------------------------
// GRU (reset_after, relu) + Dense on MI455X (gfx1250, wave32, WMMA bf16).
//   B=64, T=256, IN=256, U=768 (3U=2304), OUT=64
// Strategy:
//   1) swizzle f32 weights -> per-wave bf16 B-fragment images (coalesced loads)
//      swizzle x           -> per-wave bf16 A-fragment images (once; kills the
//                             144x-redundant f32->bf16 conversion in phase 1)
//   2) xp = x@kernel + b_in : big WMMA GEMM (pure b128-load + WMMA inner loop)
//   3) 256 sequential step kernels: inner = h@rec_kernel (+b_rec) via WMMA,
//      on-the-fly A conversion (amortized 3x over z/r/h, co-executes with XDL),
//      fused sigmoid/relu gate math, double-buffered h
//   4) out = h_last@w_out + b_out via WMMA
// ---------------------------------------------------------------------------

typedef __attribute__((ext_vector_type(16))) __bf16 bf16x16;
typedef __attribute__((ext_vector_type(8)))  float  f32x8;

#define WMMA_BF16(A, Bm, C) \
    __builtin_amdgcn_wmma_f32_16x16x32_bf16(false, (A), false, (Bm), (short)0, (C), false, false)

// --- A fragment (16x32 bf16) from a row-major f32 matrix --------------------
// 16-bit A layout (wave32): lane&15 = row M; lane>>4 selects K-offset group.
// Per lane the 16 halves are two contiguous 8-element K chunks:
//   j=0..7  -> K = kBase + g*8 + j
//   j=8..15 -> K = kBase + 16 + g*8 + (j-8)
__device__ __forceinline__ bf16x16
load_a_frag(const float* __restrict__ A, int ldA, int rowBase, int kBase, int lane) {
    const int row = rowBase + (lane & 15);
    const int g   = lane >> 4;
    const float* p = A + (size_t)row * ldA + kBase + g * 8;
    float4 c0 = *(const float4*)(p + 0);
    float4 c1 = *(const float4*)(p + 4);
    float4 c2 = *(const float4*)(p + 16);
    float4 c3 = *(const float4*)(p + 20);
    bf16x16 a;
    a[0]  = (__bf16)c0.x; a[1]  = (__bf16)c0.y; a[2]  = (__bf16)c0.z; a[3]  = (__bf16)c0.w;
    a[4]  = (__bf16)c1.x; a[5]  = (__bf16)c1.y; a[6]  = (__bf16)c1.z; a[7]  = (__bf16)c1.w;
    a[8]  = (__bf16)c2.x; a[9]  = (__bf16)c2.y; a[10] = (__bf16)c2.z; a[11] = (__bf16)c2.w;
    a[12] = (__bf16)c3.x; a[13] = (__bf16)c3.y; a[14] = (__bf16)c3.z; a[15] = (__bf16)c3.w;
    return a;
}

// --- A swizzle: f32 [M,K] row-major -> bf16 A-fragment images ---------------
// Fragment (mt,kt): fragIdx = mt*KT + kt; per-lane image exactly matches what
// the WMMA A operand expects, so the GEMM loads it as one v16bf (2x b128).
__global__ void swizzle_a(const float* __restrict__ A, __bf16* __restrict__ out,
                          int ldA, int KT) {
    const int wid  = (blockIdx.x * blockDim.x + threadIdx.x) >> 5;
    const int lane = threadIdx.x & 31;
    const int kt = wid % KT;
    const int mt = wid / KT;
    bf16x16 v = load_a_frag(A, ldA, mt * 16, kt * 32, lane);
    ((bf16x16*)out)[(size_t)wid * 32 + lane] = v;
}

// --- weight swizzle: f32 [K,N] row-major -> bf16 B-fragment images ----------
// Fragment (nt,kt): fragIdx = nt*KT + kt. Within a fragment: lane holds the
// 16 halves for column n = nt*16 + (lane&15); group g = lane>>4 covers
// K = kt*32 + g*8 + j (j<8), kt*32 + 16 + g*8 + (j-8) (j>=8)  (matches A layout).
__global__ void swizzle_w(const float* __restrict__ W, __bf16* __restrict__ out,
                          int N, int KT) {
    const int wid  = (blockIdx.x * blockDim.x + threadIdx.x) >> 5;
    const int lane = threadIdx.x & 31;
    const int kt = wid % KT;
    const int nt = wid / KT;
    const int g  = lane >> 4;
    const int n  = nt * 16 + (lane & 15);
    const int kbase = kt * 32 + g * 8;
    bf16x16 v;
#pragma unroll
    for (int j = 0; j < 16; ++j)
        v[j] = (__bf16)W[(size_t)(kbase + (j < 8 ? j : j + 8)) * N + n];
    ((bf16x16*)out)[(size_t)wid * 32 + lane] = v;
}

// --- phase 1: xp = x @ kernel + b_in  (M=16384, K=256, N=2304) --------------
__global__ void xproj_gemm(const __bf16* __restrict__ swzX, const __bf16* __restrict__ swzK,
                           const float* __restrict__ b_in, float* __restrict__ xp) {
    const int KT = 8;                                  // 256/32
    const int wid  = (blockIdx.x * blockDim.x + threadIdx.x) >> 5;
    const int lane = threadIdx.x & 31;
    const int mt = wid & 1023;                         // 1024 M tiles
    const int nt = wid >> 10;                          // 144 N tiles
    const bf16x16* Af = (const bf16x16*)swzX;
    const bf16x16* Bf = (const bf16x16*)swzK;
    f32x8 acc = {};
#pragma unroll
    for (int kk = 0; kk < KT; ++kk) {
        bf16x16 a = Af[(size_t)(mt * KT + kk) * 32 + lane];
        bf16x16 b = Bf[(size_t)(nt * KT + kk) * 32 + lane];
        acc = WMMA_BF16(a, b, acc);
    }
    const int col = nt * 16 + (lane & 15);
    const int g   = lane >> 4;
    const float bv = b_in[col];
#pragma unroll
    for (int i = 0; i < 8; ++i) {
        const int row = mt * 16 + i + g * 8;
        xp[(size_t)row * 2304 + col] = acc[i] + bv;
    }
}

// --- phase 2: one GRU step --------------------------------------------------
// inner = h @ rec_kernel + b_rec ; z=sig(xz+iz); r=sig(xr+ir);
// hh = relu(xh + r*ih); h' = z*h + (1-z)*hh
__global__ void gru_step(const float* __restrict__ hin, float* __restrict__ hout,
                         const float* __restrict__ xp, const __bf16* __restrict__ swzR,
                         const float* __restrict__ b_rec, int t) {
    const int KT = 24;                                 // 768/32
    const int wid  = (blockIdx.x * blockDim.x + threadIdx.x) >> 5;
    const int lane = threadIdx.x & 31;
    const int mt = wid & 3;                            // 4 M tiles (64 rows)
    const int nt = wid >> 2;                           // 48 N tiles (768 cols)
    const bf16x16* Bf = (const bf16x16*)swzR;
    f32x8 az = {}, ar = {}, ah = {};
    for (int kk = 0; kk < KT; ++kk) {
        bf16x16 a  = load_a_frag(hin, 768, mt * 16, kk * 32, lane);
        bf16x16 bz = Bf[(size_t)((nt      ) * KT + kk) * 32 + lane];
        bf16x16 br = Bf[(size_t)((nt +  48) * KT + kk) * 32 + lane];
        bf16x16 bh = Bf[(size_t)((nt +  96) * KT + kk) * 32 + lane];
        if (kk + 1 < KT)
            __builtin_prefetch(&Bf[(size_t)(nt * KT + kk + 1) * 32 + lane], 0, 1);
        az = WMMA_BF16(a, bz, az);
        ar = WMMA_BF16(a, br, ar);
        ah = WMMA_BF16(a, bh, ah);
    }
    const int col = nt * 16 + (lane & 15);
    const int g   = lane >> 4;
    const float bz_ = b_rec[col];
    const float br_ = b_rec[col + 768];
    const float bh_ = b_rec[col + 1536];
#pragma unroll
    for (int i = 0; i < 8; ++i) {
        const int b = mt * 16 + i + g * 8;             // batch index
        const float* xrow = xp + ((size_t)b * 256 + t) * 2304;
        const float xz = xrow[col];
        const float xr = xrow[col + 768];
        const float xh = xrow[col + 1536];
        const float iz = az[i] + bz_;
        const float ir = ar[i] + br_;
        const float ih = ah[i] + bh_;
        const float z  = 1.0f / (1.0f + __expf(-(xz + iz)));
        const float r  = 1.0f / (1.0f + __expf(-(xr + ir)));
        float hh = xh + r * ih;
        hh = hh > 0.0f ? hh : 0.0f;
        const float hp = hin[(size_t)b * 768 + col];
        hout[(size_t)b * 768 + col] = z * hp + (1.0f - z) * hh;
    }
}

// --- phase 3: out = h_last @ w_out + b_out  (64 x 64, K=768) ----------------
__global__ void dense_out(const float* __restrict__ h, const __bf16* __restrict__ swzW,
                          const float* __restrict__ b_out, float* __restrict__ out) {
    const int KT = 24;
    const int wid  = (blockIdx.x * blockDim.x + threadIdx.x) >> 5;
    const int lane = threadIdx.x & 31;
    const int mt = wid & 3;
    const int nt = wid >> 2;
    const bf16x16* Bf = (const bf16x16*)swzW;
    f32x8 acc = {};
    for (int kk = 0; kk < KT; ++kk) {
        bf16x16 a = load_a_frag(h, 768, mt * 16, kk * 32, lane);
        bf16x16 b = Bf[(size_t)(nt * KT + kk) * 32 + lane];
        acc = WMMA_BF16(a, b, acc);
    }
    const int col = nt * 16 + (lane & 15);
    const int g   = lane >> 4;
    const float bv = b_out[col];
#pragma unroll
    for (int i = 0; i < 8; ++i) {
        const int row = mt * 16 + i + g * 8;
        out[(size_t)row * 64 + col] = acc[i] + bv;
    }
}

__global__ void zero_f32(float* __restrict__ p, int n) {
    int i = blockIdx.x * blockDim.x + threadIdx.x;
    if (i < n) p[i] = 0.0f;
}

// ---------------------------------------------------------------------------
extern "C" void kernel_launch(void* const* d_in, const int* in_sizes, int n_in,
                              void* d_out, int out_size, void* d_ws, size_t ws_size,
                              hipStream_t stream) {
    const float* x      = (const float*)d_in[0];   // [64,256,256]
    const float* kernelW= (const float*)d_in[1];   // [256,2304]
    const float* recW   = (const float*)d_in[2];   // [768,2304]
    const float* bias   = (const float*)d_in[3];   // [2,2304]
    const float* w_out  = (const float*)d_in[4];   // [768,64]
    const float* b_out  = (const float*)d_in[5];   // [64]
    float* out = (float*)d_out;                    // [64,64]

    // workspace layout (all sizes multiples of 256B)
    char* w = (char*)d_ws;
    size_t off = 0;
    float*  xp   = (float*)(w + off);  off += (size_t)64 * 256 * 2304 * 4;  // 151.0 MB
    __bf16* swzX = (__bf16*)(w + off); off += (size_t)16384 * 256 * 2;      // x A-frags (8.4 MB)
    __bf16* swzK = (__bf16*)(w + off); off += (size_t)256 * 2304 * 2;       // kernel
    __bf16* swzR = (__bf16*)(w + off); off += (size_t)768 * 2304 * 2;       // rec_kernel
    __bf16* swzW = (__bf16*)(w + off); off += (size_t)768 * 64 * 2;         // w_out
    float*  h0   = (float*)(w + off);  off += (size_t)64 * 768 * 4;
    float*  h1   = (float*)(w + off);  off += (size_t)64 * 768 * 4;
    float*  hbuf[2] = { h0, h1 };

    // 1) swizzle weights to bf16 B-fragment images (one wave per 32x16 frag)
    //    kernel:     KT=8,  NT=144 -> 1152 frags
    //    rec_kernel: KT=24, NT=144 -> 3456 frags
    //    w_out:      KT=24, NT=4   ->   96 frags
    swizzle_w<<<1152 / 4, 128, 0, stream>>>(kernelW, swzK, 2304, 8);
    swizzle_w<<<3456 / 4, 128, 0, stream>>>(recW,    swzR, 2304, 24);
    swizzle_w<<<96   / 4, 128, 0, stream>>>(w_out,   swzW, 64,   24);
    //    x -> bf16 A-fragment images: MT=1024, KT=8 -> 8192 frags
    swizzle_a<<<8192 / 4, 128, 0, stream>>>(x, swzX, 256, 8);

    // 2) h0 = 0
    zero_f32<<<(64 * 768 + 255) / 256, 256, 0, stream>>>(h0, 64 * 768);

    // 3) xp = x @ kernel + b_in  (147456 tiles, 8 waves/block)
    xproj_gemm<<<147456 / 8, 256, 0, stream>>>(swzX, swzK, bias, xp);

    // 4) sequential scan: 256 dependent steps (192 waves each)
    for (int t = 0; t < 256; ++t) {
        gru_step<<<48, 128, 0, stream>>>(hbuf[t & 1], hbuf[(t + 1) & 1],
                                         xp, swzR, bias + 2304, t);
    }
    // after step 255 the latest h is in hbuf[0]

    // 5) dense head
    dense_out<<<4, 128, 0, stream>>>(hbuf[0], swzW, b_out, out);
}
